// MultiHeadAttention_28604482191447
// MI455X (gfx1250) — matrix operand order repaired and checked by reading, not compiled, not run on hardware
//
#include <hip/hip_runtime.h>
#include <hip/hip_bf16.h>

// ---------------- problem constants (fixed by the reference) ----------------
#define DIMN   1024
#define HEADS  16
#define HDN    64
#define BATCH  2
#define LQN    2048
#define LKN    2048
#define MTOT   (BATCH * LQN)      // 4096 rows for the projection GEMMs
#define SCALEF 0.125f             // 64^-0.5

// ---------------- types ----------------
typedef __attribute__((ext_vector_type(16))) __bf16 v16bf;
typedef __attribute__((ext_vector_type(8)))  float  v8f;
typedef __attribute__((ext_vector_type(4)))  float  f32x4;
typedef __attribute__((ext_vector_type(4)))  unsigned int u32x4;

union Frag {
    v16bf          v;
    unsigned short u[16];
    u32x4          q[2];
};

__device__ __forceinline__ unsigned short f2bf(float f) {
    union { float f; unsigned int u; } x;
    x.f = f;
    unsigned int r = x.u + 0x7FFFu + ((x.u >> 16) & 1u);   // round-to-nearest-even
    return (unsigned short)(r >> 16);
}

// =====================================================================
// One-shot fp32 -> bf16 conversion (memory-bound, runs once per tensor;
// removes all conversion VALU from the GEMM hot loops).
// n must be a multiple of 8.
// =====================================================================
__global__ __launch_bounds__(256)
void cvt_f32_bf16_kernel(const float* __restrict__ in,
                         unsigned short* __restrict__ out, int n)
{
    const int i = (blockIdx.x * 256 + threadIdx.x) * 8;
    if (i >= n) return;
    f32x4 a = *(const f32x4*)(in + i);
    f32x4 b = *(const f32x4*)(in + i + 4);
    u32x4 o;
    o.x = (unsigned)f2bf(a.x) | ((unsigned)f2bf(a.y) << 16);
    o.y = (unsigned)f2bf(a.z) | ((unsigned)f2bf(a.w) << 16);
    o.z = (unsigned)f2bf(b.x) | ((unsigned)f2bf(b.y) << 16);
    o.w = (unsigned)f2bf(b.z) | ((unsigned)f2bf(b.w) << 16);
    *(u32x4*)(out + i) = o;
}

// =====================================================================
// GEMM: Y[m,n] = sum_k X[m,k] * W[n,k] + bias[n]   (torch Linear)
// X, W already bf16.  M = 4096, N = K = 1024.
// Block tile 128x64, 8 waves, wave tile 32x32 (2x2 WMMA).
// Double-buffered LDS, one barrier per k-step; staging is pure b128 copies.
// mode 0: out bf16 [B,H,L,Hd]      (Q, K projections)
// mode 1: out bf16 [B,H,Hd,L]      (V projection, transposed)
// mode 2: out f32  [M,N]           (output projection / final result)
// =====================================================================
__global__ __launch_bounds__(256)
void mha_gemm_kernel(const unsigned short* __restrict__ X,
                     const unsigned short* __restrict__ W,
                     const float* __restrict__ bias,
                     void* __restrict__ outp, int mode)
{
    __shared__ unsigned short Xs[2][128 * 40];  // 128 rows x 32 k (stride 40)
    __shared__ unsigned short Wsh[2][64 * 40];  // 64 rows  x 32 k (stride 40)

    const int tid  = threadIdx.x;
    const int lane = tid & 31;
    const int wave = tid >> 5;
    const int wm   = wave & 3;    // wave M index 0..3
    const int wn   = wave >> 2;   // wave N index 0..1
    const int half = lane >> 4;
    const int ln   = lane & 15;
    const int m0   = blockIdx.y * 128;
    const int n0   = blockIdx.x * 64;

    // staging map: X row xr gets cols [xc,xc+16); W row wr gets cols [wc,wc+8)
    const int xr = tid >> 1, xc = (tid & 1) * 16;
    const int wr = tid >> 2, wc = (tid & 3) * 8;
    const unsigned short* Xsrc = X + (size_t)(m0 + xr) * DIMN + xc;
    const unsigned short* Wsrc = W + (size_t)(n0 + wr) * DIMN + wc;

    v8f acc[2][2];
#pragma unroll
    for (int i = 0; i < 2; ++i)
#pragma unroll
        for (int j = 0; j < 2; ++j)
            acc[i][j] = (v8f){0.f,0.f,0.f,0.f,0.f,0.f,0.f,0.f};

    // ---- preload tile 0 into buffer 0 ----
    {
        unsigned short* xd = &Xs[0][xr * 40 + xc];
        *(u32x4*)xd       = *(const u32x4*)Xsrc;
        *(u32x4*)(xd + 8) = *(const u32x4*)(Xsrc + 8);
        unsigned short* wd = &Wsh[0][wr * 40 + wc];
        *(u32x4*)wd       = *(const u32x4*)Wsrc;
    }

    for (int kt = 0; kt < DIMN / 32; ++kt) {
        const int  cur  = kt & 1;
        const bool more = (kt + 1) < (DIMN / 32);
        __syncthreads();

        // prefetch next tile from global into registers (overlaps compute)
        u32x4 xa0, xa1, wa0;
        if (more) {
            const int k0 = (kt + 1) * 32;
            xa0 = *(const u32x4*)(Xsrc + k0);
            xa1 = *(const u32x4*)(Xsrc + k0 + 8);
            wa0 = *(const u32x4*)(Wsrc + k0);
        }

        // A fragments: a[j] = X[row][ j<8 ? 8*half+j : 16+8*half+(j-8) ]
        Frag Af[2];
#pragma unroll
        for (int i = 0; i < 2; ++i) {
            const unsigned short* p = &Xs[cur][(wm * 32 + i * 16 + ln) * 40 + 8 * half];
            Af[i].q[0] = *(const u32x4*)p;
            Af[i].q[1] = *(const u32x4*)(p + 16);
        }
        // B fragments: b[j] = W[col][ j<8 ? 8*half+j : 16+8*half+(j-8) ]
        Frag Bf[2];
#pragma unroll
        for (int i = 0; i < 2; ++i) {
            const unsigned short* p = &Wsh[cur][(wn * 32 + i * 16 + ln) * 40 + 8 * half];
            Bf[i].q[0] = *(const u32x4*)p;
            Bf[i].q[1] = *(const u32x4*)(p + 16);
        }
#pragma unroll
        for (int mi = 0; mi < 2; ++mi)
#pragma unroll
            for (int ni = 0; ni < 2; ++ni)
                acc[mi][ni] = __builtin_amdgcn_wmma_f32_16x16x32_bf16(
                    false, Af[mi].v, false, Bf[ni].v,
                    (short)0, acc[mi][ni], false, false);

        // store prefetched tile into the other buffer
        if (more) {
            unsigned short* xd = &Xs[cur ^ 1][xr * 40 + xc];
            *(u32x4*)xd       = xa0;
            *(u32x4*)(xd + 8) = xa1;
            unsigned short* wd = &Wsh[cur ^ 1][wr * 40 + wc];
            *(u32x4*)wd       = wa0;
        }
    }

    // epilogue
#pragma unroll
    for (int mi = 0; mi < 2; ++mi) {
#pragma unroll
        for (int ni = 0; ni < 2; ++ni) {
            const int gn = n0 + wn * 32 + ni * 16 + ln;
            const float bv = bias[gn];
#pragma unroll
            for (int r = 0; r < 8; ++r) {
                const float c  = acc[mi][ni][r] + bv;
                const int   gm = m0 + wm * 32 + mi * 16 + r + 8 * half;
                if (mode == 2) {
                    ((float*)outp)[(size_t)gm * DIMN + gn] = c;
                } else {
                    const int bb = gm >> 11;        // / LQN
                    const int l  = gm & 2047;
                    const int h  = gn >> 6;         // / HDN
                    const int d  = gn & 63;
                    size_t idx;
                    if (mode == 0)
                        idx = (((size_t)bb * HEADS + h) * LQN + l) * HDN + d;
                    else
                        idx = (((size_t)bb * HEADS + h) * HDN + d) * LKN + l;
                    ((unsigned short*)outp)[idx] = f2bf(c);
                }
            }
        }
    }
}

// =====================================================================
// Flash attention: block = 8 waves = 128 query rows of one (b,h).
// K/V tiles (64 keys) staged once per block into double-buffered LDS and
// shared by all 8 waves.  Computes S^T = K * Q^T so the exp(S^T)
// D-fragment is exactly the A-fragment of P for P @ V (no shuffles).
// Qh [B,H,LQ,Hd] bf16 ; Kh [B,H,LK,Hd] bf16 ; Vt [B,H,Hd,LK] bf16
// Ao [B,LQ,DIM] bf16 (consumed by the final projection GEMM)
// =====================================================================
__global__ __launch_bounds__(256)
void mha_attn_kernel(const unsigned short* __restrict__ Qh,
                     const unsigned short* __restrict__ Kh,
                     const unsigned short* __restrict__ Vt,
                     const int* __restrict__ kvm,
                     unsigned short* __restrict__ Ao)
{
    __shared__ unsigned short Ks[2][64 * 72];   // [key][d]  (stride 72)
    __shared__ unsigned short Vs[2][64 * 72];   // [d][key]  (stride 72)

    const int tid  = threadIdx.x;
    const int lane = tid & 31;
    const int wave = tid >> 5;
    const int half = lane >> 4;
    const int ln   = lane & 15;

    const int bh   = blockIdx.x >> 4;          // / (LQ/128)
    const int qblk = blockIdx.x & 15;
    const int b    = bh >> 4;                  // / HEADS
    const int h    = bh & 15;
    const int q0   = qblk * 128 + wave * 16;

    const unsigned short* Qb = Qh + ((size_t)bh * LQN + q0) * HDN;
    const unsigned short* Kb = Kh + (size_t)bh * LKN * HDN;
    const unsigned short* Vb = Vt + (size_t)bh * HDN * LKN;
    const int* mb = kvm + (size_t)b * LKN;

    // staging map: row sr (key for K / d for V), 16 bf16 cols at sc
    const int sr = tid >> 2;
    const int sc = (tid & 3) * 16;

    // Q B-fragments (held for whole loop): b[j] = Q[q0+ln][i*32 + (j<8 ? 8*half+j : 16+8*half+(j-8))]
    Frag QB[2];
#pragma unroll
    for (int i = 0; i < 2; ++i) {
        const unsigned short* p = Qb + ln * HDN + i * 32 + 8 * half;
        QB[i].q[0] = *(const u32x4*)p;
        QB[i].q[1] = *(const u32x4*)(p + 16);
    }

    float mrun = -1e30f, lrun = 0.f;
    v8f oacc[4];
#pragma unroll
    for (int t = 0; t < 4; ++t)
        oacc[t] = (v8f){0.f,0.f,0.f,0.f,0.f,0.f,0.f,0.f};

    // ---- preload key-tile 0 into buffer 0 ----
    {
        const unsigned short* kp = Kb + (size_t)sr * HDN + sc;
        unsigned short* kd = &Ks[0][sr * 72 + sc];
        *(u32x4*)kd       = *(const u32x4*)kp;
        *(u32x4*)(kd + 8) = *(const u32x4*)(kp + 8);
        const unsigned short* vp = Vb + (size_t)sr * LKN + sc;
        unsigned short* vd = &Vs[0][sr * 72 + sc];
        *(u32x4*)vd       = *(const u32x4*)vp;
        *(u32x4*)(vd + 8) = *(const u32x4*)(vp + 8);
    }

    for (int kt = 0; kt < LKN / 64; ++kt) {
        const int  cur  = kt & 1;
        const bool more = (kt + 1) < (LKN / 64);
        __syncthreads();

        // prefetch next 64-key tile into registers
        u32x4 ka0, ka1, va0, va1;
        if (more) {
            const int kk = (kt + 1) * 64;
            const unsigned short* kp = Kb + (size_t)(kk + sr) * HDN + sc;
            ka0 = *(const u32x4*)kp;
            ka1 = *(const u32x4*)(kp + 8);
            const unsigned short* vp = Vb + (size_t)sr * LKN + kk + sc;
            va0 = *(const u32x4*)vp;
            va1 = *(const u32x4*)(vp + 8);
        }

        const int kkbase = kt * 64;
#pragma unroll
        for (int inner = 0; inner < 2; ++inner) {
            const int klo = inner * 32;

            // K A-fragments from LDS
            Frag KA[2][2];
#pragma unroll
            for (int sub = 0; sub < 2; ++sub)
#pragma unroll
                for (int i = 0; i < 2; ++i) {
                    const unsigned short* p =
                        &Ks[cur][(klo + sub * 16 + ln) * 72 + i * 32 + 8 * half];
                    KA[sub][i].q[0] = *(const u32x4*)p;
                    KA[sub][i].q[1] = *(const u32x4*)(p + 16);
                }

            v8f st[2];
#pragma unroll
            for (int sub = 0; sub < 2; ++sub) {
                v8f z = (v8f){0.f,0.f,0.f,0.f,0.f,0.f,0.f,0.f};
                z = __builtin_amdgcn_wmma_f32_16x16x32_bf16(
                        false, KA[sub][0].v, false, QB[0].v, (short)0, z, false, false);
                st[sub] = __builtin_amdgcn_wmma_f32_16x16x32_bf16(
                        false, KA[sub][1].v, false, QB[1].v, (short)0, z, false, false);
            }

            // scale + mask + tile row(=query) max
            float tmax = -1e30f;
#pragma unroll
            for (int sub = 0; sub < 2; ++sub)
#pragma unroll
                for (int r = 0; r < 8; ++r) {
                    float s = st[sub][r] * SCALEF;
                    const int key = kkbase + klo + sub * 16 + r + 8 * half;
                    s = mb[key] ? s : -1e30f;
                    st[sub][r] = s;
                    tmax = fmaxf(tmax, s);
                }
            tmax = fmaxf(tmax, __shfl_xor(tmax, 16, 32));
            const float mnew  = fmaxf(mrun, tmax);
            const float alpha = __expf(mrun - mnew);

            float lsum = 0.f;
            Frag PA;
#pragma unroll
            for (int sub = 0; sub < 2; ++sub)
#pragma unroll
                for (int r = 0; r < 8; ++r) {
                    const float p = __expf(st[sub][r] - mnew);
                    lsum += p;
                    PA.u[sub * 8 + r] = f2bf(p);   // == A-fragment layout of P
                }
            lsum += __shfl_xor(lsum, 16, 32);
            lrun = lrun * alpha + lsum;
            mrun = mnew;

            // per-slot rescale: query of slot r is (r + 8*half), stats in that lane
            float ascale[8];
#pragma unroll
            for (int r = 0; r < 8; ++r)
                ascale[r] = __shfl(alpha, r + 8 * half, 32);

#pragma unroll
            for (int t = 0; t < 4; ++t) {
                Frag VB;   // b[j] = V[key=klo+(j<8 ? 8*half+j : 16+8*half+(j-8))][d=t*16+ln]
                const unsigned short* p =
                    &Vs[cur][(t * 16 + ln) * 72 + klo + 8 * half];
                VB.q[0] = *(const u32x4*)p;
                VB.q[1] = *(const u32x4*)(p + 16);
                v8f a = oacc[t];
#pragma unroll
                for (int r = 0; r < 8; ++r)
                    a[r] *= ascale[r];
                oacc[t] = __builtin_amdgcn_wmma_f32_16x16x32_bf16(
                              false, PA.v, false, VB.v, (short)0, a, false, false);
            }
        }

        // store prefetched tile into the other buffer
        if (more) {
            unsigned short* kd = &Ks[cur ^ 1][sr * 72 + sc];
            *(u32x4*)kd       = ka0;
            *(u32x4*)(kd + 8) = ka1;
            unsigned short* vd = &Vs[cur ^ 1][sr * 72 + sc];
            *(u32x4*)vd       = va0;
            *(u32x4*)(vd + 8) = va1;
        }
    }

    float linv[8];
#pragma unroll
    for (int r = 0; r < 8; ++r) {
        float l = __shfl(lrun, r + 8 * half, 32);
        linv[r] = 1.f / fmaxf(l, 1e-20f);
    }
#pragma unroll
    for (int t = 0; t < 4; ++t)
#pragma unroll
        for (int r = 0; r < 8; ++r) {
            const int q = q0 + r + 8 * half;
            Ao[((size_t)b * LQN + q) * DIMN + h * HDN + t * 16 + ln] =
                f2bf(oacc[t][r] * linv[r]);
        }
}

// =====================================================================
extern "C" void kernel_launch(void* const* d_in, const int* in_sizes, int n_in,
                              void* d_out, int out_size, void* d_ws, size_t ws_size,
                              hipStream_t stream)
{
    (void)in_sizes; (void)n_in; (void)out_size; (void)ws_size;
    const float* q  = (const float*)d_in[0];
    const float* k  = (const float*)d_in[1];
    const float* v  = (const float*)d_in[2];
    const int*   km = (const int*)  d_in[3];
    const float* Wq = (const float*)d_in[4];
    const float* bq = (const float*)d_in[5];
    const float* Wk = (const float*)d_in[6];
    const float* bk = (const float*)d_in[7];
    const float* Wv = (const float*)d_in[8];
    const float* bv = (const float*)d_in[9];
    const float* Wo = (const float*)d_in[10];
    const float* bo = (const float*)d_in[11];

    const size_t nAct = (size_t)MTOT * DIMN;                 // 4M elems (q/k/v/Ao)
    const size_t nWgt = (size_t)DIMN * DIMN;                 // 1M elems per weight
    unsigned short* p   = (unsigned short*)d_ws;
    unsigned short* qbf = p;  p += nAct;
    unsigned short* kbf = p;  p += nAct;
    unsigned short* vbf = p;  p += nAct;
    unsigned short* wqb = p;  p += nWgt;
    unsigned short* wkb = p;  p += nWgt;
    unsigned short* wvb = p;  p += nWgt;
    unsigned short* wob = p;  p += nWgt;
    unsigned short* Qh  = p;  p += nAct;   // [B,H,LQ,Hd]
    unsigned short* Kh  = p;  p += nAct;   // [B,H,LK,Hd]
    unsigned short* Vt  = p;  p += nAct;   // [B,H,Hd,LK]
    unsigned short* Ao  = p;  p += nAct;   // [B,LQ,DIM]

    // one-shot bf16 conversions (memory-bound)
    const int gAct = (int)(nAct / 8 / 256);   // 2048 blocks
    const int gWgt = (int)(nWgt / 8 / 256);   // 512 blocks
    cvt_f32_bf16_kernel<<<gAct, 256, 0, stream>>>(q,  qbf, (int)nAct);
    cvt_f32_bf16_kernel<<<gAct, 256, 0, stream>>>(k,  kbf, (int)nAct);
    cvt_f32_bf16_kernel<<<gAct, 256, 0, stream>>>(v,  vbf, (int)nAct);
    cvt_f32_bf16_kernel<<<gWgt, 256, 0, stream>>>(Wq, wqb, (int)nWgt);
    cvt_f32_bf16_kernel<<<gWgt, 256, 0, stream>>>(Wk, wkb, (int)nWgt);
    cvt_f32_bf16_kernel<<<gWgt, 256, 0, stream>>>(Wv, wvb, (int)nWgt);
    cvt_f32_bf16_kernel<<<gWgt, 256, 0, stream>>>(Wo, wob, (int)nWgt);

    dim3 gg(DIMN / 64, MTOT / 128);   // (16, 32)
    mha_gemm_kernel<<<gg, 256, 0, stream>>>(qbf, wqb, bq, (void*)Qh, 0);
    mha_gemm_kernel<<<gg, 256, 0, stream>>>(kbf, wkb, bk, (void*)Kh, 0);
    mha_gemm_kernel<<<gg, 256, 0, stream>>>(vbf, wvb, bv, (void*)Vt, 1);

    mha_attn_kernel<<<BATCH * HEADS * (LQN / 128), 256, 0, stream>>>(Qh, Kh, Vt, km, Ao);

    mha_gemm_kernel<<<gg, 256, 0, stream>>>(Ao, wob, bo, d_out, 2);
}
